// GCN_3685081940548
// MI455X (gfx1250) — matrix operand order repaired and checked by reading, not compiled, not run on hardware
//
#include <hip/hip_runtime.h>
#include <hip/hip_bf16.h>

// ---------------- types / helpers ----------------
typedef __attribute__((ext_vector_type(16))) __bf16 v16bf;
typedef __attribute__((ext_vector_type(8)))  float  v8f;
typedef __attribute__((ext_vector_type(4)))  unsigned int u32x4;

#define NEG_FILL (-9.0e15f)

union Frag { v16bf v; u32x4 q[2]; };

__device__ __forceinline__ unsigned short f2bf(float x) {
    unsigned u = __float_as_uint(x);
    u += 0x7FFFu + ((u >> 16) & 1u);           // round-to-nearest-even
    return (unsigned short)(u >> 16);
}

__device__ __forceinline__ v8f wmma_bf16(v16bf a, v16bf b, v8f c) {
    return __builtin_amdgcn_wmma_f32_16x16x32_bf16(false, a, false, b, (short)0, c, false, false);
}

// A fragment (16x32 bf16): row = base + (lane&15); chunks K: [d, d+7] and [d+16, d+23]
// with d = kbase + (lane>=16 ? 8 : 0)   (matches ISA 16-bit A layout)
__device__ __forceinline__ v16bf load_frag_a(const unsigned short* __restrict__ src,
                                             int ld, int row, int kbase) {
    int lane = threadIdx.x & 31;
    const unsigned short* p = src + (size_t)(row + (lane & 15)) * ld
                                  + kbase + ((lane & 16) ? 8 : 0);
    Frag f;
    f.q[0] = *(const u32x4*)(p);
    f.q[1] = *(const u32x4*)(p + 16);
    return f.v;
}

// B fragment (32x16 bf16) from Bt stored row-major (N x K): col = colbase + (lane&15);
// chunks K: [k, k+7] and [k+16, k+23] with k = kbase + (lane>=16 ? 8 : 0)
__device__ __forceinline__ v16bf load_frag_b(const unsigned short* __restrict__ bt,
                                             int ld, int col, int kbase) {
    int lane = threadIdx.x & 31;
    const unsigned short* p = bt + (size_t)(col + (lane & 15)) * ld
                                 + kbase + ((lane & 16) ? 8 : 0);
    Frag f;
    f.q[0] = *(const u32x4*)(p);
    f.q[1] = *(const u32x4*)(p + 16);
    return f.v;
}

// ---------------- f32 -> bf16 conversion ----------------
__global__ void k_cvt_bf16(const float* __restrict__ in, unsigned short* __restrict__ out, long n) {
    long i = (long)blockIdx.x * blockDim.x + threadIdx.x;
    long stride = (long)gridDim.x * blockDim.x;
    for (; i < n; i += stride) out[i] = f2bf(in[i]);
}

// ---------------- bf16 tiled transpose (R x C) -> (C x R), batched over z ----------------
__global__ void k_transpose_bf16(const unsigned short* __restrict__ in,
                                 unsigned short* __restrict__ out, int R, int C) {
    __shared__ unsigned short tile[32][33];
    const unsigned short* src = in  + (size_t)blockIdx.z * R * C;
    unsigned short*       dst = out + (size_t)blockIdx.z * R * C;
    int r0 = blockIdx.y * 32, c0 = blockIdx.x * 32;
    int tx = threadIdx.x & 31, ty = threadIdx.x >> 5;   // 256 threads: ty 0..7
    #pragma unroll
    for (int i = 0; i < 32; i += 8)
        tile[ty + i][tx] = src[(size_t)(r0 + ty + i) * C + c0 + tx];
    __syncthreads();
    #pragma unroll
    for (int i = 0; i < 32; i += 8)
        dst[(size_t)(c0 + ty + i) * R + r0 + tx] = tile[tx][ty + i];
}

// ---------------- generic NT GEMM: C(Mx256) = A(MxK) x Bt(256xK)^T, bf16 in, f32 acc ----
// Block tile: 64 rows x 256 cols, 8 waves. Wave: 2 m-frags x 4 n-frags = 8 WMMAs/K-step.
// 2-stage ping-pong software pipeline over K (step 64): no buffer-rotation copies.
// Requires K % 64 == 0 (true for all call sites: 512, 4096, 256).
// epilogue: 0=none, 1=prelu(alpha), 2=bias+prelu(alpha)
__global__ __launch_bounds__(256, 1)
void k_gemm_nt(const unsigned short* __restrict__ A, long sA, int lda,
               const unsigned short* __restrict__ Bt, long sB, int ldb,
               void* __restrict__ C, long sC, int ldc,
               int K, int c_is_f32, int epilogue,
               const float* __restrict__ alpha, const float* __restrict__ bias) {
    int wave = threadIdx.x >> 5, lane = threadIdx.x & 31;
    int msub = wave & 1, ng = wave >> 1;               // 2 x 32-row halves, 4 x 64-col groups
    int m0 = blockIdx.x * 64 + msub * 32;
    const unsigned short* Ab  = A  + sA * blockIdx.y;
    const unsigned short* Btb = Bt + sB * blockIdx.y;

    v16bf afA[2], bfA[4], afB[2], bfB[4];
    afA[0] = load_frag_a(Ab, lda, m0, 0);
    afA[1] = load_frag_a(Ab, lda, m0 + 16, 0);
    #pragma unroll
    for (int i = 0; i < 4; i++) bfA[i] = load_frag_b(Btb, ldb, ng * 64 + i * 16, 0);

    v8f acc[8] = {};
    for (int k = 0; k < K; k += 64) {
        // stage 1: issue loads for k+32, consume buffers for k
        afB[0] = load_frag_a(Ab, lda, m0, k + 32);
        afB[1] = load_frag_a(Ab, lda, m0 + 16, k + 32);
        #pragma unroll
        for (int i = 0; i < 4; i++) bfB[i] = load_frag_b(Btb, ldb, ng * 64 + i * 16, k + 32);
        __builtin_prefetch(Ab + (size_t)(m0 + (lane & 15)) * lda + k + 64, 0, 1);
        #pragma unroll
        for (int i = 0; i < 4; i++) {
            acc[i]     = wmma_bf16(afA[0], bfA[i], acc[i]);
            acc[4 + i] = wmma_bf16(afA[1], bfA[i], acc[4 + i]);
        }
        // stage 2: issue loads for k+64 (if any), consume buffers for k+32
        if (k + 64 < K) {
            afA[0] = load_frag_a(Ab, lda, m0, k + 64);
            afA[1] = load_frag_a(Ab, lda, m0 + 16, k + 64);
            #pragma unroll
            for (int i = 0; i < 4; i++) bfA[i] = load_frag_b(Btb, ldb, ng * 64 + i * 16, k + 64);
        }
        #pragma unroll
        for (int i = 0; i < 4; i++) {
            acc[i]     = wmma_bf16(afB[0], bfB[i], acc[i]);
            acc[4 + i] = wmma_bf16(afB[1], bfB[i], acc[4 + i]);
        }
    }
    #pragma unroll
    for (int ms = 0; ms < 2; ms++) {
        #pragma unroll
        for (int i = 0; i < 4; i++) {
            int col = ng * 64 + i * 16 + (lane & 15);
            #pragma unroll
            for (int j = 0; j < 8; j++) {
                float v = acc[ms * 4 + i][j];
                int m = m0 + ms * 16 + j + 8 * (lane >> 4);
                if (epilogue == 1) { float a = *alpha; v = (v >= 0.f) ? v : a * v; }
                else if (epilogue == 2) { v += bias[col]; float a = *alpha; v = (v >= 0.f) ? v : a * v; }
                size_t idx = (size_t)sC * blockIdx.y + (size_t)m * ldc + col;
                if (c_is_f32) ((float*)C)[idx] = v;
                else          ((unsigned short*)C)[idx] = f2bf(v);
            }
        }
    }
}

// ---------------- Q/K projection + L2 normalize + div-loss partials ----------------
__global__ __launch_bounds__(256, 1)
void k_qkproj(const unsigned short* __restrict__ out_bf,
              const unsigned short* __restrict__ wq, const unsigned short* __restrict__ wk,
              unsigned short* __restrict__ q_bf, unsigned short* __restrict__ k_bf,
              float* __restrict__ divp, int N) {
    __shared__ float qraw[16][256];
    __shared__ float kraw[16][256];
    int b = blockIdx.y, mbase = blockIdx.x * 16;
    const unsigned short* X = out_bf + (size_t)b * N * 256;
    int wave = threadIdx.x >> 5, lane = threadIdx.x & 31;
    int isK = (wave >= 4), g = wave & 3;
    const unsigned short* W = isK ? wk : wq;

    v8f acc[4] = {};
    #pragma unroll
    for (int k = 0; k < 256; k += 32) {        // constant trip count: fully unrolled, SSA-renamed
        v16bf af = load_frag_a(X, 256, mbase, k);
        #pragma unroll
        for (int i = 0; i < 4; i++) {
            v16bf bf = load_frag_b(W, 256, g * 64 + i * 16, k);
            acc[i] = wmma_bf16(af, bf, acc[i]);
        }
    }
    float (*dst)[256] = isK ? kraw : qraw;
    #pragma unroll
    for (int i = 0; i < 4; i++) {
        int col = g * 64 + i * 16 + (lane & 15);
        #pragma unroll
        for (int j = 0; j < 8; j++) dst[j + 8 * (lane >> 4)][col] = acc[i][j];
    }
    __syncthreads();
    if (threadIdx.x < 16) {
        int r = threadIdx.x, row = mbase + r;
        float sq = 0.f, sk = 0.f;
        for (int f = 0; f < 256; f++) { float q = qraw[r][f], k = kraw[r][f]; sq += q * q; sk += k * k; }
        float rq = 1.f / fmaxf(sqrtf(sq), 1e-12f);
        float rk = 1.f / fmaxf(sqrtf(sk), 1e-12f);
        float d = 0.f;
        size_t base = (size_t)b * N * 256 + (size_t)row * 256;
        for (int f = 0; f < 256; f++) {
            float qn = qraw[r][f] * rq, kn = kraw[r][f] * rk;
            q_bf[base + f] = f2bf(qn);
            k_bf[base + f] = f2bf(kn);
            float t = qn - kn; d += t * t;
        }
        divp[(size_t)b * N + row] = d;
    }
}

// ---------------- attention pass A: masked-softmax row max / row sum ----------------
__global__ __launch_bounds__(256, 1)
void k_attn_stats(const unsigned short* __restrict__ q_bf, const unsigned short* __restrict__ k_bf,
                  const unsigned short* __restrict__ adj_bf,
                  float* __restrict__ rowmax, float* __restrict__ rowsum, int N) {
    __shared__ float stage[8][16][32];
    __shared__ float wmax[8][16];
    __shared__ float wsum[8][16];
    int b = blockIdx.y, mbase = blockIdx.x * 16;
    int wave = threadIdx.x >> 5, lane = threadIdx.x & 31;
    const unsigned short* Q   = q_bf  + (size_t)b * N * 256;
    const unsigned short* Km  = k_bf  + (size_t)b * N * 256;
    const unsigned short* Adj = adj_bf + (size_t)b * N * N;

    v16bf qf[8];
    #pragma unroll
    for (int kk = 0; kk < 8; kk++) qf[kk] = load_frag_a(Q, 256, mbase, kk * 32);

    float runM = -3.0e38f, runS = 0.f;      // lanes 0..15 carry row stats
    for (int nb = wave * 32; nb < N; nb += 256) {
        #pragma unroll
        for (int t = 0; t < 2; t++) {
            int ncol = nb + t * 16;
            int nn = ncol + (lane & 15);
            // preload mask bits for this tile
            unsigned short am[8];
            #pragma unroll
            for (int j = 0; j < 8; j++)
                am[j] = Adj[(size_t)(mbase + j + 8 * (lane >> 4)) * N + nn];
            // pipelined S accumulation (fully unrolled: no rotation copies)
            v8f s = {};
            v16bf b0 = load_frag_b(Km, 256, ncol, 0);
            #pragma unroll
            for (int kk = 0; kk < 8; kk++) {
                v16bf b1 = (kk < 7) ? load_frag_b(Km, 256, ncol, (kk + 1) * 32) : b0;
                s = wmma_bf16(qf[kk], b0, s);
                b0 = b1;
            }
            #pragma unroll
            for (int j = 0; j < 8; j++) {
                int m = j + 8 * (lane >> 4);
                stage[wave][m][(lane & 15) + t * 16] = am[j] ? s[j] : NEG_FILL;
            }
        }
        if (lane < 16) {
            float tmax = -3.0e38f;
            #pragma unroll
            for (int c = 0; c < 32; c++) tmax = fmaxf(tmax, stage[wave][lane][c]);
            float nm = fmaxf(runM, tmax);
            float ss = 0.f;
            #pragma unroll
            for (int c = 0; c < 32; c++) ss += __expf(stage[wave][lane][c] - nm);
            runS = runS * __expf(runM - nm) + ss;
            runM = nm;
        }
    }
    if (lane < 16) { wmax[wave][lane] = runM; wsum[wave][lane] = runS; }
    __syncthreads();
    if (threadIdx.x < 16) {
        int r = threadIdx.x;
        float M = -3.0e38f;
        for (int w = 0; w < 8; w++) M = fmaxf(M, wmax[w][r]);
        float S = 0.f;
        for (int w = 0; w < 8; w++) S += wsum[w][r] * __expf(wmax[w][r] - M);
        rowmax[(size_t)b * N + mbase + r] = M;
        rowsum[(size_t)b * N + mbase + r] = S;
    }
}

// ---------------- attention pass B: ctx = softmax(scores) @ out ----------------
__global__ __launch_bounds__(256, 1)
void k_attn_ctx(const unsigned short* __restrict__ q_bf, const unsigned short* __restrict__ k_bf,
                const unsigned short* __restrict__ adj_bf, const unsigned short* __restrict__ outT_bf,
                const float* __restrict__ rowmax, const float* __restrict__ rowsum,
                unsigned short* __restrict__ ctx_bf, int N) {
    __shared__ unsigned short pstage[8][16][32];
    __shared__ float ctxbuf[16][256];
    __shared__ float rmax_s[16], rsum_s[16];
    int b = blockIdx.y, mbase = blockIdx.x * 16;
    int wave = threadIdx.x >> 5, lane = threadIdx.x & 31;
    if (threadIdx.x < 16) {
        rmax_s[threadIdx.x] = rowmax[(size_t)b * N + mbase + threadIdx.x];
        rsum_s[threadIdx.x] = rowsum[(size_t)b * N + mbase + threadIdx.x];
    }
    for (int i = threadIdx.x; i < 16 * 256; i += 256) ((float*)ctxbuf)[i] = 0.f;
    __syncthreads();

    const unsigned short* Q   = q_bf  + (size_t)b * N * 256;
    const unsigned short* Km  = k_bf  + (size_t)b * N * 256;
    const unsigned short* Adj = adj_bf + (size_t)b * N * N;
    const unsigned short* VT  = outT_bf + (size_t)b * 256 * N;

    v16bf qf[8];
    #pragma unroll
    for (int kk = 0; kk < 8; kk++) qf[kk] = load_frag_a(Q, 256, mbase, kk * 32);

    v8f acc[16] = {};     // 16 feature tiles of 16x16 ctx partials per wave
    for (int nb = wave * 32; nb < N; nb += 256) {
        #pragma unroll
        for (int t = 0; t < 2; t++) {
            int ncol = nb + t * 16;
            int nn = ncol + (lane & 15);
            unsigned short am[8];
            #pragma unroll
            for (int j = 0; j < 8; j++)
                am[j] = Adj[(size_t)(mbase + j + 8 * (lane >> 4)) * N + nn];
            v8f s = {};
            v16bf b0 = load_frag_b(Km, 256, ncol, 0);
            #pragma unroll
            for (int kk = 0; kk < 8; kk++) {
                v16bf b1 = (kk < 7) ? load_frag_b(Km, 256, ncol, (kk + 1) * 32) : b0;
                s = wmma_bf16(qf[kk], b0, s);
                b0 = b1;
            }
            #pragma unroll
            for (int j = 0; j < 8; j++) {
                int m = j + 8 * (lane >> 4);
                float v = am[j] ? s[j] : NEG_FILL;
                pstage[wave][m][(lane & 15) + t * 16] = f2bf(__expf(v - rmax_s[m]));
            }
        }
        // P back out of LDS in A layout (per-wave private staging; LDS ops are in-order)
        v16bf pf;
        {
            const unsigned short* p = &pstage[wave][lane & 15][(lane & 16) ? 8 : 0];
            Frag f;
            f.q[0] = *(const u32x4*)(p);
            f.q[1] = *(const u32x4*)(p + 16);
            pf = f.v;
        }
        // pipelined P@V over 16 feature tiles
        auto load_vt = [&](int ft) -> v16bf {
            int f = ft * 16 + (lane & 15);
            const unsigned short* p = VT + (size_t)f * N + nb + ((lane & 16) ? 8 : 0);
            Frag fr;
            fr.q[0] = *(const u32x4*)(p);
            fr.q[1] = *(const u32x4*)(p + 16);
            return fr.v;
        };
        v16bf vb = load_vt(0);
        #pragma unroll
        for (int ft = 0; ft < 16; ft++) {
            v16bf vn = (ft < 15) ? load_vt(ft + 1) : vb;
            acc[ft] = wmma_bf16(pf, vb, acc[ft]);
            vb = vn;
        }
    }
    // reduce partial ctx across waves
    #pragma unroll
    for (int ft = 0; ft < 16; ft++) {
        int col = ft * 16 + (lane & 15);
        #pragma unroll
        for (int j = 0; j < 8; j++)
            atomicAdd(&ctxbuf[j + 8 * (lane >> 4)][col], acc[ft][j]);
    }
    __syncthreads();
    {
        int r = threadIdx.x >> 4;
        int c0 = (threadIdx.x & 15) * 16;
        float rs = 1.f / rsum_s[r];
        size_t base = (size_t)b * N * 256 + (size_t)(mbase + r) * 256 + c0;
        #pragma unroll
        for (int c = 0; c < 16; c++) ctx_bf[base + c] = f2bf(ctxbuf[r][c0 + c] * rs);
    }
}

// ---------------- div-loss mean reduction ----------------
__global__ void k_divloss(const float* __restrict__ divp, float* __restrict__ out, int n) {
    __shared__ float sh[256];
    float s = 0.f;
    for (int i = threadIdx.x; i < n; i += 256) s += divp[i];
    sh[threadIdx.x] = s;
    __syncthreads();
    for (int st = 128; st > 0; st >>= 1) {
        if (threadIdx.x < st) sh[threadIdx.x] += sh[threadIdx.x + st];
        __syncthreads();
    }
    if (threadIdx.x == 0) out[0] = sh[0] / (float)n;
}

// ---------------- host launcher ----------------
extern "C" void kernel_launch(void* const* d_in, const int* in_sizes, int n_in,
                              void* d_out, int out_size, void* d_ws, size_t ws_size,
                              hipStream_t stream) {
    (void)in_sizes; (void)n_in; (void)out_size; (void)ws_size;
    const int Bn = 2, N = 4096, INF = 512, OUTF = 256;

    const float* seq  = (const float*)d_in[0];
    const float* adj  = (const float*)d_in[1];
    const float* Wfc  = (const float*)d_in[2];
    const float* Wq   = (const float*)d_in[3];
    const float* Wk   = (const float*)d_in[4];
    const float* Wv1  = (const float*)d_in[5];
    const float* Wv2  = (const float*)d_in[6];
    const float* a_v  = (const float*)d_in[7];
    const float* a_ac = (const float*)d_in[8];
    const float* bias = (const float*)d_in[9];

    char* W = (char*)d_ws;
    size_t off = 0;
    auto take = [&](size_t bytes) { size_t p = off; off += (bytes + 255) & ~(size_t)255; return p; };

    unsigned short* adj_bf = (unsigned short*)(W + take((size_t)Bn * N * N * 2));
    unsigned short* seq_bf = (unsigned short*)(W + take((size_t)Bn * N * INF * 2));
    unsigned short* wfc_bf = (unsigned short*)(W + take((size_t)OUTF * INF * 2));
    unsigned short* wq_bf  = (unsigned short*)(W + take((size_t)OUTF * OUTF * 2));
    unsigned short* wk_bf  = (unsigned short*)(W + take((size_t)OUTF * OUTF * 2));
    unsigned short* wv1_bf = (unsigned short*)(W + take((size_t)OUTF * OUTF * 2));
    unsigned short* wv2_bf = (unsigned short*)(W + take((size_t)OUTF * OUTF * 2));
    unsigned short* sf_bf  = (unsigned short*)(W + take((size_t)Bn * N * OUTF * 2));
    unsigned short* sfT_bf = (unsigned short*)(W + take((size_t)Bn * N * OUTF * 2));
    unsigned short* out_bf = (unsigned short*)(W + take((size_t)Bn * N * OUTF * 2));
    unsigned short* outT_bf= (unsigned short*)(W + take((size_t)Bn * N * OUTF * 2));
    unsigned short* q_bf   = (unsigned short*)(W + take((size_t)Bn * N * OUTF * 2));
    unsigned short* k_bf   = (unsigned short*)(W + take((size_t)Bn * N * OUTF * 2));
    unsigned short* ctx_bf = (unsigned short*)(W + take((size_t)Bn * N * OUTF * 2));
    unsigned short* h_bf   = (unsigned short*)(W + take((size_t)Bn * N * OUTF * 2));
    float* rowmax = (float*)(W + take((size_t)Bn * N * 4));
    float* rowsum = (float*)(W + take((size_t)Bn * N * 4));
    float* divp   = (float*)(W + take((size_t)Bn * N * 4));

    // 1) conversions
    k_cvt_bf16<<<2048, 256, 0, stream>>>(adj,  adj_bf, (long)Bn * N * N);
    k_cvt_bf16<<<1024, 256, 0, stream>>>(seq,  seq_bf, (long)Bn * N * INF);
    k_cvt_bf16<<<64,   256, 0, stream>>>(Wfc,  wfc_bf, (long)OUTF * INF);
    k_cvt_bf16<<<32,   256, 0, stream>>>(Wq,   wq_bf,  (long)OUTF * OUTF);
    k_cvt_bf16<<<32,   256, 0, stream>>>(Wk,   wk_bf,  (long)OUTF * OUTF);
    k_cvt_bf16<<<32,   256, 0, stream>>>(Wv1,  wv1_bf, (long)OUTF * OUTF);
    k_cvt_bf16<<<32,   256, 0, stream>>>(Wv2,  wv2_bf, (long)OUTF * OUTF);

    // 2) seq_fts = seq @ Wfc^T  (M = B*N = 8192, K = 512); 64-row block tiles
    k_gemm_nt<<<dim3(128, 1), 256, 0, stream>>>(seq_bf, 0, INF, wfc_bf, 0, INF,
                                                sf_bf, 0, OUTF, INF, 0, 0, nullptr, nullptr);
    // 3) seq_fts^T per batch (N x 256 -> 256 x N)
    k_transpose_bf16<<<dim3(OUTF / 32, N / 32, Bn), 256, 0, stream>>>(sf_bf, sfT_bf, N, OUTF);
    // 4) out = adj @ seq_fts  (per batch: M = 4096, K = 4096) in NT form against sfT
    k_gemm_nt<<<dim3(64, Bn), 256, 0, stream>>>(adj_bf, (long)N * N, N,
                                                sfT_bf, (long)OUTF * N, N,
                                                out_bf, (long)N * OUTF, OUTF,
                                                N, 0, 0, nullptr, nullptr);
    // 5) Q/K projection + normalize + div-loss partials
    k_qkproj<<<dim3(N / 16, Bn), 256, 0, stream>>>(out_bf, wq_bf, wk_bf, q_bf, k_bf, divp, N);
    // 6) out^T per batch (V in feature-major for P@V)
    k_transpose_bf16<<<dim3(OUTF / 32, N / 32, Bn), 256, 0, stream>>>(out_bf, outT_bf, N, OUTF);
    // 7) masked softmax stats
    k_attn_stats<<<dim3(N / 16, Bn), 256, 0, stream>>>(q_bf, k_bf, adj_bf, rowmax, rowsum, N);
    // 8) ctx = softmax(scores) @ out
    k_attn_ctx<<<dim3(N / 16, Bn), 256, 0, stream>>>(q_bf, k_bf, adj_bf, outT_bf,
                                                     rowmax, rowsum, ctx_bf, N);
    // 9) h = prelu(ctx @ Wv1^T, a_v)
    k_gemm_nt<<<dim3(128, 1), 256, 0, stream>>>(ctx_bf, 0, OUTF, wv1_bf, 0, OUTF,
                                                h_bf, 0, OUTF, OUTF, 0, 1, a_v, nullptr);
    // 10) y = prelu(h @ Wv2^T + bias, a_act)  -> f32 into d_out
    k_gemm_nt<<<dim3(128, 1), 256, 0, stream>>>(h_bf, 0, OUTF, wv2_bf, 0, OUTF,
                                                d_out, 0, OUTF, OUTF, 1, 2, a_ac, bias);
    // 11) div_loss = mean(divp)
    k_divloss<<<1, 256, 0, stream>>>(divp, (float*)d_out + (size_t)Bn * N * OUTF, Bn * N);
}
